// BF_NIR_conv_33079838113947
// MI455X (gfx1250) — matrix operand order repaired and checked by reading, compiled, not run on hardware
//
#include <hip/hip_runtime.h>
#include <hip/hip_bf16.h>

typedef __attribute__((ext_vector_type(16))) __bf16 v16bf;
typedef __attribute__((ext_vector_type(8)))  float  v8f;
typedef unsigned int u32x4 __attribute__((ext_vector_type(4)));
typedef int          i32x4 __attribute__((ext_vector_type(4)));
typedef int          i32x8 __attribute__((ext_vector_type(8)));

#if __has_include(<hip/amd_detail/amd_gfx1250_TDM.h>)
#define TDM_6ARG 1
#endif

union FragBF {
    v16bf v;
    uint4 q[2];
};

#define NPIX  65536                // 256 x 256 HR pixels
#define TILE_PIX 16
#define NROWS 64                   // TILE_PIX * 4 branches
#define K1PAD 416                  // 386 padded to 13*32
#define K1CH  13
#define N1    256
#define K2CH  8
#define N2    128
#define K3CH  4
#define N3    32

// packed-weight sizes in bf16 elements (one 16x16x32 B-tile = 512 halfwords)
#define W1P_SZ (K1CH * (N1 / 16) * 512)   // 106496
#define W2P_SZ (K2CH * (N2 / 16) * 512)   // 32768
#define W3P_SZ (K3CH * (N3 / 16) * 512)   // 4096
#define W2P_OFF W1P_SZ
#define W3P_OFF (W1P_SZ + W2P_SZ)

// LDS plan (dynamic, 119040 B; aliased across barrier-separated phases)
#define ATILE_OFF 0                // 64 x 416 bf16 = 53248  (fill + L1 + gcs)
#define ACT1_OFF  0                // 64 x 256 bf16 = 32768  (after L1 barrier)
#define ACT2_OFF  32768            // 64 x 128 bf16 = 16384
#define PREDS_OFF 49152            // 64 x 32  f32  = 8192   (layer3; ring dead)
#define WBUF_OFF  53248            // 4 x 16384 B TDM weight ring buffer
#define GCS_OFF   118784           // 64 f32 (never overlapped)
#define SMEM_TOTAL 119040

// ---------------------------------------------------------------------------
// 16-bit fragment order used here for A and B (contiguous k): lane l holds 16 packed halfwords
// j=0..15 with K = (l<16 ? 0 : 16) + j, N(or M) = l&15.
// ---------------------------------------------------------------------------
__device__ __forceinline__ void frag_coord(int within, int& kOff, int& n) {
    int l = within >> 4;
    int j = within & 15;
    int base = (l < 16) ? 0 : 16;
    kOff = base + j;
    n = l & 15;
}

__device__ __forceinline__ unsigned int lds_addr_of(const void* p) {
    return (unsigned int)(uintptr_t)(__attribute__((address_space(3))) const void*)p;
}

__device__ __forceinline__ void frag_load(FragBF& f, const __bf16* p) {
    f.q[0] = *(const uint4*)p;
    f.q[1] = *(const uint4*)(p + 8);
}

// ---------------------------------------------------------------------------
// TDM: flat 1D DMA of (n8 * 8) bytes from global to LDS (ISA ch.8 descriptor,
// 2D form: data_size=8B, tile_dim0=tensor_dim0=stride=n8, tile_dim1=1).
// Issue from ONE wave only; track with TENSORcnt.
// ---------------------------------------------------------------------------
__device__ __forceinline__ void tdm_load_1d(const void* gsrc,
                                            unsigned int lds_byte_addr,
                                            unsigned int n8) {
    unsigned long long ga = (unsigned long long)(uintptr_t)gsrc;
    u32x4 g0;
    g0.x = 0x1u;                                       // count=1, user mode
    g0.y = lds_byte_addr;                              // lds_addr
    g0.z = (unsigned int)(ga & 0xFFFFFFFFu);           // global_addr[31:0]
    g0.w = (unsigned int)((ga >> 32) & 0x01FFFFFFu)    // global_addr[56:32]
         | (2u << 30);                                 // type = 2 ("image")
    i32x8 g1;
    g1[0] = (3 << 16);                                 // data_size = 8 bytes
    g1[1] = (int)((n8 & 0xFFFFu) << 16);               // tensor_dim0[15:0]
    g1[2] = (int)(((n8 >> 16) & 0xFFFFu)               // tensor_dim0[31:16]
         | (1u << 16));                                // tensor_dim1 = 1
    g1[3] = (int)((n8 & 0xFFFFu) << 16);               // tile_dim0
    g1[4] = 1;                                         // tile_dim1=1, tile_dim2=0
    g1[5] = (int)n8;                                   // tensor_dim0_stride lo32
    g1[6] = 0;
    g1[7] = 0;
    i32x4 z4 = {0, 0, 0, 0};
#ifdef TDM_6ARG
    i32x8 z8 = {0, 0, 0, 0, 0, 0, 0, 0};
    __builtin_amdgcn_tensor_load_to_lds(g0, g1, z4, z4, z8, 0);
#else
    __builtin_amdgcn_tensor_load_to_lds(g0, g1, z4, z4, 0);
#endif
}

// ---------------------------------------------------------------------------
// Kernel 1: pack W1/W2/W3 (f32 row-major) into bf16 WMMA B-fragment order.
// ---------------------------------------------------------------------------
__global__ void pack_weights_kernel(const float* __restrict__ W1,
                                    const float* __restrict__ W2,
                                    const float* __restrict__ W3,
                                    __bf16* __restrict__ ws) {
    int o = blockIdx.x * blockDim.x + threadIdx.x;

    if (o < W1P_SZ) {
        int tile = o >> 9, within = o & 511;
        int kt = tile >> 4, nt = tile & 15;   // 13 k-chunks x 16 n-tiles
        int ko, n;
        frag_coord(within, ko, n);
        int K = kt * 32 + ko;
        int N = nt * 16 + n;
        float v = (K < 386) ? W1[K * 256 + N] : 0.0f;
        ws[o] = (__bf16)v;
    }
    if (o < W2P_SZ) {
        int tile = o >> 9, within = o & 511;
        int kt = tile >> 3, nt = tile & 7;    // 8 k-chunks x 8 n-tiles
        int ko, n;
        frag_coord(within, ko, n);
        ws[W2P_OFF + o] = (__bf16)W2[(kt * 32 + ko) * 128 + nt * 16 + n];
    }
    if (o < W3P_SZ) {
        int tile = o >> 9, within = o & 511;
        int kt = tile >> 1, nt = tile & 1;    // 4 k-chunks x 2 n-tiles
        int ko, n;
        frag_coord(within, ko, n);
        ws[W3P_OFF + o] = (__bf16)W3[(kt * 32 + ko) * 32 + nt * 16 + n];
    }
}

// ---------------------------------------------------------------------------
// Kernel 2: fused gather + 3-layer bf16-WMMA MLP + softmax blend.
// One block (256 threads = 8 wave32) handles 16 HR pixels (64 MLP rows).
// W1/W2 B-tiles stream through a 4-deep TDM ring in LDS (lookahead 2, one
// barrier per chunk); the inner product loop is a 2-stage software pipeline
// (load tile nt+1 while WMMA consumes tile nt).
// ---------------------------------------------------------------------------
__global__ void __launch_bounds__(256)
fused_mlp_kernel(const float* __restrict__ feat,
                 const float* __restrict__ lr_guide,
                 const float* __restrict__ hr_guide,
                 const float* __restrict__ b1,
                 const float* __restrict__ b2,
                 const float* __restrict__ b3,
                 const __bf16* __restrict__ w1p,
                 const __bf16* __restrict__ w2p,
                 const __bf16* __restrict__ w3p,
                 float* __restrict__ out) {
    extern __shared__ __align__(16) char smem_raw[];
    __bf16* a_tile = (__bf16*)(smem_raw + ATILE_OFF);
    __bf16* act1   = (__bf16*)(smem_raw + ACT1_OFF);
    __bf16* act2   = (__bf16*)(smem_raw + ACT2_OFF);
    float*  preds  = (float*)(smem_raw + PREDS_OFF);
    __bf16* wbuf   = (__bf16*)(smem_raw + WBUF_OFF);
    float*  gcs    = (float*)(smem_raw + GCS_OFF);

    const int tid  = threadIdx.x;
    const int pix0 = blockIdx.x * TILE_PIX;
    const int w    = tid >> 5;       // wave id (wave32)
    const int l    = tid & 31;
    const unsigned int wb = lds_addr_of(wbuf);

    // kick off DMA of W1 chunks 0 and 1 while we gather the input tile
    if (w == 0) {
        tdm_load_1d(w1p, wb, 2048);                       // 16 KB -> buf0
        tdm_load_1d(w1p + 8192, wb + 16384, 2048);        // 16 KB -> buf1
    }

    // ------------------- Phase 0: gather input tile (bf16) -----------------
    {
        int r    = tid >> 2;            // row 0..63 (4 threads per row)
        int k0   = (tid & 3) * 104;     // 4 * 104 = 416
        int pl   = r >> 2;
        int br   = r & 3;
        int p    = pix0 + pl;
        int Y    = p >> 8;
        int X    = p & 255;
        float cy = (2.0f * Y + 1.0f) / 256.0f - 1.0f;
        float cx = (2.0f * X + 1.0f) / 256.0f - 1.0f;
        float vx = (br & 2) ? 1.0f : -1.0f;
        float vy = (br & 1) ? 1.0f : -1.0f;
        float sy = cy + vx * (1.0f / 128.0f);
        float sx = cx + vy * (1.0f / 128.0f);
        int iy   = (int)rintf(((sy + 1.0f) * 128.0f - 1.0f) * 0.5f);
        int ix   = (int)rintf(((sx + 1.0f) * 128.0f - 1.0f) * 0.5f);
        bool valid = (iy >= 0) && (iy < 128) && (ix >= 0) && (ix < 128);
        int iyc = min(max(iy, 0), 127);
        int ixc = min(max(ix, 0), 127);
        float qcy = valid ? (2.0f * iyc + 1.0f) / 128.0f - 1.0f : 0.0f;
        float qcx = valid ? (2.0f * ixc + 1.0f) / 128.0f - 1.0f : 0.0f;
        int sp = iyc * 128 + ixc;

        for (int k = k0; k < k0 + 104; ++k) {
            float v;
            if (k < 128)       v = valid ? lr_guide[k * 16384 + sp] : 0.0f;
            else if (k < 256)  v = valid ? feat[(k - 128) * 16384 + sp] : 0.0f;
            else if (k < 384)  v = hr_guide[(k - 256) * NPIX + p];
            else if (k == 384) v = (cy - qcy) * 128.0f;
            else if (k == 385) v = (cx - qcx) * 128.0f;
            else               v = 0.0f;
            a_tile[r * K1PAD + k] = (__bf16)v;
        }
    }
    __syncthreads();

    // gcs: dot of featc channels [252..254] at center sample vs branch sample
    if (tid < NROWS) {
        int r  = tid;
        int pl = r >> 2;
        int p  = pix0 + pl;
        int Y  = p >> 8;
        int X  = p & 255;
        float cy = (2.0f * Y + 1.0f) / 256.0f - 1.0f;
        float cx = (2.0f * X + 1.0f) / 256.0f - 1.0f;
        int iy = (int)rintf(((cy + 1.0f) * 128.0f - 1.0f) * 0.5f);
        int ix = (int)rintf(((cx + 1.0f) * 128.0f - 1.0f) * 0.5f);
        iy = min(max(iy, 0), 127);
        ix = min(max(ix, 0), 127);
        int sp = iy * 128 + ix;
        float g = 0.0f;
        #pragma unroll
        for (int j = 0; j < 3; ++j)
            g += feat[(124 + j) * 16384 + sp] * (float)a_tile[r * K1PAD + 252 + j];
        gcs[r] = g;
    }

    // wave decomposition: 8 waves = 4 row-stripes x 2 column halves
    const int stripe  = w >> 1;
    const int half    = w & 1;
    const int rowBase = stripe * 16;
    const int lane15  = l & 15;
    const int laneHi  = l >> 4;          // 0 or 1
    const int kbase   = laneHi ? 16 : 0;  // 16-bit fragment K offset

    // ------------------- Layer 1: [64 x 416] @ [416 x 256] ------------------
    v8f acc1[8];
    #pragma unroll
    for (int nt = 0; nt < 8; ++nt)
        #pragma unroll
        for (int i = 0; i < 8; ++i) acc1[nt][i] = 0.0f;

    #pragma unroll
    for (int kt = 0; kt < K1CH; ++kt) {
        if (w == 0) {
            if (kt + 2 < K1CH) {
                tdm_load_1d(w1p + (kt + 2) * 8192,
                            wb + (((kt + 2) & 3) << 14), 2048);
                __builtin_amdgcn_s_wait_tensorcnt(2);   // chunk kt has landed
            } else if (kt + 2 == K1CH) {
                __builtin_amdgcn_s_wait_tensorcnt(1);
            } else {
                __builtin_amdgcn_s_wait_tensorcnt(0);
            }
        }
        __syncthreads();   // publish chunk kt; also fences ring reuse (dist 2)

        FragBF af;
        frag_load(af, a_tile + (rowBase + lane15) * K1PAD + kt * 32 + kbase);

        const __bf16* wp = wbuf + ((kt & 3) << 13) + (half * 8) * 512 + l * 16;

        // 2-stage software pipeline: load nt+1 while WMMA consumes nt
        FragBF cur, nxt;
        frag_load(cur, wp);
        #pragma unroll
        for (int nt = 0; nt < 8; ++nt) {
            if (nt + 1 < 8)
                frag_load(nxt, wp + (nt + 1) * 512);
            acc1[nt] = __builtin_amdgcn_wmma_f32_16x16x32_bf16(
                false, af.v, false, cur.v, (short)0, acc1[nt], false, false);
            if (nt + 1 < 8)
                cur = nxt;
        }
    }
    __syncthreads();   // all W1 ring + a_tile reads complete

    // stage W2 chunks 0,1 (ring buffers 0,1 are dead behind the barrier)
    if (w == 0) {
        tdm_load_1d(w2p, wb, 1024);                       // 8 KB -> buf0
        tdm_load_1d(w2p + 4096, wb + 16384, 1024);        // 8 KB -> buf1
    }

    {   // bias + relu + store act1 (aliases dead a_tile region)
        int colBase = half * 128;
        #pragma unroll
        for (int nt = 0; nt < 8; ++nt) {
            int col = colBase + nt * 16 + lane15;
            float bias = b1[col];
            #pragma unroll
            for (int v = 0; v < 8; ++v) {
                float x = acc1[nt][v] + bias;
                x = x > 0.0f ? x : 0.0f;
                act1[(rowBase + v + laneHi * 8) * 256 + col] = (__bf16)x;
            }
        }
    }
    // warm W3 into the caches for layer 3
    __builtin_prefetch(w3p + tid * 16, 0, 0);

    // ------------------- Layer 2: [64 x 256] @ [256 x 128] ------------------
    v8f acc2[4];
    #pragma unroll
    for (int nt = 0; nt < 4; ++nt)
        #pragma unroll
        for (int i = 0; i < 8; ++i) acc2[nt][i] = 0.0f;

    #pragma unroll
    for (int kt = 0; kt < K2CH; ++kt) {
        if (w == 0) {
            if (kt + 2 < K2CH) {
                tdm_load_1d(w2p + (kt + 2) * 4096,
                            wb + (((kt + 2) & 3) << 14), 1024);
                __builtin_amdgcn_s_wait_tensorcnt(2);
            } else if (kt + 2 == K2CH) {
                __builtin_amdgcn_s_wait_tensorcnt(1);
            } else {
                __builtin_amdgcn_s_wait_tensorcnt(0);
            }
        }
        __syncthreads();   // publish chunk kt; first iter also publishes act1

        FragBF af;
        frag_load(af, act1 + (rowBase + lane15) * 256 + kt * 32 + kbase);

        const __bf16* wp = wbuf + ((kt & 3) << 13) + (half * 4) * 512 + l * 16;

        FragBF cur, nxt;
        frag_load(cur, wp);
        #pragma unroll
        for (int nt = 0; nt < 4; ++nt) {
            if (nt + 1 < 4)
                frag_load(nxt, wp + (nt + 1) * 512);
            acc2[nt] = __builtin_amdgcn_wmma_f32_16x16x32_bf16(
                false, af.v, false, cur.v, (short)0, acc2[nt], false, false);
            if (nt + 1 < 4)
                cur = nxt;
        }
    }
    __syncthreads();   // all W2 ring reads complete

    {
        int colBase = half * 64;
        #pragma unroll
        for (int nt = 0; nt < 4; ++nt) {
            int col = colBase + nt * 16 + lane15;
            float bias = b2[col];
            #pragma unroll
            for (int v = 0; v < 8; ++v) {
                float x = acc2[nt][v] + bias;
                x = x > 0.0f ? x : 0.0f;
                act2[(rowBase + v + laneHi * 8) * 128 + col] = (__bf16)x;
            }
        }
    }
    __syncthreads();

    // ------------------- Layer 3: [64 x 128] @ [128 x 32] -------------------
    if (half == 0) {
        v8f acc3[2];
        #pragma unroll
        for (int nt = 0; nt < 2; ++nt)
            #pragma unroll
            for (int i = 0; i < 8; ++i) acc3[nt][i] = 0.0f;

        #pragma unroll
        for (int kt = 0; kt < K3CH; ++kt) {
            FragBF af;
            frag_load(af, act2 + (rowBase + lane15) * 128 + kt * 32 + kbase);

            FragBF cur, nxt;
            frag_load(cur, w3p + (size_t)(kt * 2) * 512 + l * 16);
            #pragma unroll
            for (int nt = 0; nt < 2; ++nt) {
                if (nt + 1 < 2)
                    frag_load(nxt, w3p + (size_t)(kt * 2 + 1) * 512 + l * 16);
                acc3[nt] = __builtin_amdgcn_wmma_f32_16x16x32_bf16(
                    false, af.v, false, cur.v, (short)0, acc3[nt], false, false);
                if (nt + 1 < 2)
                    cur = nxt;
            }
        }

        #pragma unroll
        for (int nt = 0; nt < 2; ++nt) {
            int col = nt * 16 + lane15;
            float bias = b3[col];
            #pragma unroll
            for (int v = 0; v < 8; ++v)
                preds[(rowBase + v + laneHi * 8) * 32 + col] = acc3[nt][v] + bias;
        }
    }
    __syncthreads();

    // ------------------- Combine: softmax over 4 branches -------------------
    for (int t = tid; t < TILE_PIX * 32; t += blockDim.x) {
        int pl = t >> 5;
        int c  = t & 31;
        float g0 = gcs[pl * 4 + 0], g1 = gcs[pl * 4 + 1];
        float g2 = gcs[pl * 4 + 2], g3 = gcs[pl * 4 + 3];
        float m  = fmaxf(fmaxf(g0, g1), fmaxf(g2, g3));
        float e0 = __expf(g0 - m), e1 = __expf(g1 - m);
        float e2 = __expf(g2 - m), e3 = __expf(g3 - m);
        float inv = 1.0f / (e0 + e1 + e2 + e3);
        float o = (e0 * preds[(pl * 4 + 0) * 32 + c] +
                   e1 * preds[(pl * 4 + 1) * 32 + c] +
                   e2 * preds[(pl * 4 + 2) * 32 + c] +
                   e3 * preds[(pl * 4 + 3) * 32 + c]) * inv;
        out[c * NPIX + pix0 + pl] = o;
    }
}

// ---------------------------------------------------------------------------
extern "C" void kernel_launch(void* const* d_in, const int* in_sizes, int n_in,
                              void* d_out, int out_size, void* d_ws, size_t ws_size,
                              hipStream_t stream) {
    const float* feat     = (const float*)d_in[0];
    const float* lr_guide = (const float*)d_in[1];
    const float* hr_guide = (const float*)d_in[2];
    const float* W1       = (const float*)d_in[3];
    const float* b1       = (const float*)d_in[4];
    const float* W2       = (const float*)d_in[5];
    const float* b2       = (const float*)d_in[6];
    const float* W3       = (const float*)d_in[7];
    const float* b3       = (const float*)d_in[8];
    float*       out      = (float*)d_out;

    __bf16* wpk = (__bf16*)d_ws;

    pack_weights_kernel<<<(W1P_SZ + 255) / 256, 256, 0, stream>>>(W1, W2, W3, wpk);

    fused_mlp_kernel<<<NPIX / TILE_PIX, 256, SMEM_TOTAL, stream>>>(
        feat, lr_guide, hr_guide, b1, b2, b3,
        wpk, wpk + W2P_OFF, wpk + W3P_OFF, out);
}
